// GDIV_84980222918787
// MI455X (gfx1250) — matrix operand order repaired and checked by reading, not compiled, not run on hardware
//
#include <hip/hip_runtime.h>
#include <math.h>

// ---------------------------------------------------------------------------
// Types for CDNA5 WMMA (wave32): bf16 A/B fragments (16 x bf16 = 8 VGPRs),
// f32 accumulator (8 VGPRs).
// ---------------------------------------------------------------------------
typedef __bf16 bf16_t;
typedef __attribute__((ext_vector_type(16))) __bf16 v16bf;
typedef __attribute__((ext_vector_type(8)))  __bf16 v8bf;
typedef __attribute__((ext_vector_type(8)))  float  v8f;
typedef __attribute__((ext_vector_type(4)))  int    v4i;

// Pointer types matching the gfx1250 async-to-LDS builtin signature:
// void __builtin_amdgcn_global_load_async_to_lds_b128(v4i as(1)*, v4i as(3)*, imm, imm)
typedef __attribute__((address_space(1))) v4i global_v4i;
typedef __attribute__((address_space(3))) v4i lds_v4i;

#define NNODE 4096
#define FIN   512
#define HID   256

// GEMM tiling: block tile 128x128, staged K-depth 64 (two 32-deep WMMA steps
// per buffer -> 16 WMMAs per barrier). LDS row stride padded to 72 elements
// (144B = 36 banks): 36*r mod 64 gives 16 disjoint 4-bank groups -> b128
// fragment reads are bank-conflict free.
#define BM 128
#define BN 128
#define BK 64
#define LDK 72

// CDNA5 async global->LDS path (ASYNCcnt), with graceful fallback.
#if defined(__has_builtin)
#  if __has_builtin(__builtin_amdgcn_global_load_async_to_lds_b128)
#    define HAVE_ASYNC_LDS 1
#  endif
#endif
#ifndef HAVE_ASYNC_LDS
#  define HAVE_ASYNC_LDS 0
#endif

#if defined(__has_builtin)
#  if __has_builtin(__builtin_amdgcn_s_wait_asynccnt)
#    define WAIT_ASYNCCNT0() __builtin_amdgcn_s_wait_asynccnt(0)
#  else
#    define WAIT_ASYNCCNT0() asm volatile("s_wait_asynccnt 0" ::: "memory")
#  endif
#else
#  define WAIT_ASYNCCNT0() asm volatile("s_wait_asynccnt 0" ::: "memory")
#endif

__device__ __forceinline__ bf16_t f2bf(float f) {
    unsigned u = __float_as_uint(f);
    unsigned r = (u + 0x7FFFu + ((u >> 16) & 1u)) >> 16;  // round-nearest-even
    unsigned short s = (unsigned short)r;
    return __builtin_bit_cast(bf16_t, s);
}

// A fragment (16x32) from LDS tile. ISA layout: lane l -> row m=l&15, half=l>>4;
// VGPRs 0-3 = K[8*half..+8), VGPRs 4-7 = K[16+8*half..+8): two 16B ds reads.
__device__ __forceinline__ v16bf lds_frag_a(const bf16_t* s, int mOff, int kOff) {
    int l  = threadIdx.x & 31;
    int hi = l >> 4;
    const bf16_t* p = s + (mOff + (l & 15)) * LDK + kOff + 8 * hi;
    union { v16bf v; v8bf h[2]; } u;
    u.h[0] = *(const v8bf*)p;
    u.h[1] = *(const v8bf*)(p + 16);
    return u.v;
}

// B fragment (32x16) from LDS tile of Bt: lane l -> col n=l&15;
// VGPRs 0-3 = K[8*half..+8), VGPRs 4-7 = K[16+8*half..+8): two 16B ds reads.
__device__ __forceinline__ v16bf lds_frag_b(const bf16_t* s, int nOff, int kOff) {
    int l  = threadIdx.x & 31;
    int hi = l >> 4;
    const bf16_t* p = s + (nOff + (l & 15)) * LDK + kOff + 8 * hi;
    union { v16bf v; v8bf h[2]; } u;
    u.h[0] = *(const v8bf*)p;
    u.h[1] = *(const v8bf*)(p + 16);
    return u.v;
}

// ---------------------------------------------------------------------------
// Universal NT GEMM: C(MxN) = act( A(MxK) * Bt(NxK)^T  [+ res] )
// Block tile 128x128 (8 waves as 4Mx2N, wave tile 32x64 = 2x4 WMMA frags),
// double-buffered async LDS staging of 128x64 tiles.
// All call sites: M,N multiples of 128, K multiple of 64 (EXEC all-ones).
// act: 0=none 1=relu 2=sigmoid. Any of outF/outB/outBT may be null.
// outBT stores C transposed (N x M) bf16 to feed the next GEMM's Bt operand.
// ---------------------------------------------------------------------------
__global__ __launch_bounds__(256)
void wgemm_nt_bf16(const bf16_t* __restrict__ A, const bf16_t* __restrict__ Bt,
                   int M, int N, int K,
                   const float* __restrict__ res, int act,
                   float* __restrict__ outF, bf16_t* __restrict__ outB,
                   bf16_t* __restrict__ outBT) {
    __shared__ bf16_t sA[2][BM * LDK];
    __shared__ bf16_t sB[2][BN * LDK];

    const int wave = threadIdx.x >> 5;
    const int lane = threadIdx.x & 31;
    const int mBlk = blockIdx.y * BM;
    const int nBlk = blockIdx.x * BN;
    const int mW = (wave >> 1) * 32;   // wave offset inside block tile
    const int nW = (wave & 1) * 64;

    // ---- staging setup: all row math hoisted out of the K loop ------------
    // Tile = 128 rows x 64 elems = 1024 16B-chunks; 256 threads x 4 chunks.
    // Thread t owns rows (t>>3)+{0,32,64,96} at element offset (t&7)*8; the
    // global address advances by +BK per K step, LDS addresses are fixed.
    const int t    = threadIdx.x;
    const int srow = t >> 3;
    const int soff = (t & 7) * 8;
    const size_t rstride = (size_t)32 * K;            // 32 rows in elements
    const bf16_t* gA = A  + (size_t)(mBlk + srow) * K + soff;
    const bf16_t* gB = Bt + (size_t)(nBlk + srow) * K + soff;
    const int lofs = srow * LDK + soff;

    auto stage = [&](int b, int k0) {
        bf16_t* la = &sA[b][0] + lofs;
        bf16_t* lb = &sB[b][0] + lofs;
        const bf16_t* ga = gA + k0;
        const bf16_t* gb = gB + k0;
#pragma unroll
        for (int i = 0; i < 4; ++i) {
#if HAVE_ASYNC_LDS
            __builtin_amdgcn_global_load_async_to_lds_b128(
                (global_v4i*)ga, (lds_v4i*)la, 0, 0);
            __builtin_amdgcn_global_load_async_to_lds_b128(
                (global_v4i*)gb, (lds_v4i*)lb, 0, 0);
#else
            *(v8bf*)la = *(const v8bf*)ga;
            *(v8bf*)lb = *(const v8bf*)gb;
#endif
            ga += rstride;            gb += rstride;
            la += 32 * LDK;           lb += 32 * LDK;
        }
    };

    v8f c[2][4] = {};

    stage(0, 0);

    int buf = 0;
    for (int k0 = 0; k0 < K; k0 += BK) {
        WAIT_ASYNCCNT0();        // own async copies landed in LDS
        __syncthreads();         // ... and everyone else's too
        if (k0 + BK < K) stage(buf ^ 1, k0 + BK);

#pragma unroll
        for (int ks = 0; ks < 2; ++ks) {
            const int ko = 32 * ks;
            v16bf a0 = lds_frag_a(sA[buf], mW,      ko);
            v16bf a1 = lds_frag_a(sA[buf], mW + 16, ko);
            v16bf b0 = lds_frag_b(sB[buf], nW,      ko);
            v16bf b1 = lds_frag_b(sB[buf], nW + 16, ko);
            v16bf b2 = lds_frag_b(sB[buf], nW + 32, ko);
            v16bf b3 = lds_frag_b(sB[buf], nW + 48, ko);

            c[0][0] = __builtin_amdgcn_wmma_f32_16x16x32_bf16(false, a0, false, b0, (short)0, c[0][0], false, false);
            c[0][1] = __builtin_amdgcn_wmma_f32_16x16x32_bf16(false, a0, false, b1, (short)0, c[0][1], false, false);
            c[0][2] = __builtin_amdgcn_wmma_f32_16x16x32_bf16(false, a0, false, b2, (short)0, c[0][2], false, false);
            c[0][3] = __builtin_amdgcn_wmma_f32_16x16x32_bf16(false, a0, false, b3, (short)0, c[0][3], false, false);
            c[1][0] = __builtin_amdgcn_wmma_f32_16x16x32_bf16(false, a1, false, b0, (short)0, c[1][0], false, false);
            c[1][1] = __builtin_amdgcn_wmma_f32_16x16x32_bf16(false, a1, false, b1, (short)0, c[1][1], false, false);
            c[1][2] = __builtin_amdgcn_wmma_f32_16x16x32_bf16(false, a1, false, b2, (short)0, c[1][2], false, false);
            c[1][3] = __builtin_amdgcn_wmma_f32_16x16x32_bf16(false, a1, false, b3, (short)0, c[1][3], false, false);
        }
        buf ^= 1;
    }

    // Epilogue. C layout: lane l -> n = n0 + (l&15); VGPR v -> m = m0 + v + 8*(l>>4).
    const int col = lane & 15;
    const int hi  = lane >> 4;
    const int mBase = mBlk + mW;
    const int nBase = nBlk + nW;
    for (int i = 0; i < 2; ++i) {
        for (int j = 0; j < 4; ++j) {
            const int n = nBase + 16 * j + col;
            for (int v = 0; v < 8; ++v) {
                const int m = mBase + 16 * i + v + 8 * hi;
                float acc = c[i][j][v];
                if (res)  acc += res[(size_t)m * N + n];
                if (act == 1)      acc = fmaxf(acc, 0.0f);
                else if (act == 2) acc = 1.0f / (1.0f + __expf(-acc));
                if (outF)  outF[(size_t)m * N + n] = acc;
                if (outB)  outB[(size_t)m * N + n] = f2bf(acc);
                if (outBT) outBT[(size_t)n * M + m] = f2bf(acc);
            }
        }
    }
}

// ---------------------------------------------------------------------------
// Helper kernels
// ---------------------------------------------------------------------------
__global__ void zero_f32_kernel(float* __restrict__ p, size_t n) {
    size_t i = (size_t)blockIdx.x * blockDim.x + threadIdx.x;
    if (i < n) p[i] = 0.0f;
}

__global__ void cvt_bf16_kernel(const float* __restrict__ in, bf16_t* __restrict__ out, size_t n) {
    size_t i = (size_t)blockIdx.x * blockDim.x + threadIdx.x;
    if (i < n) out[i] = f2bf(in[i]);
}

// out[c*rows + r] = bf16(in[r*cols + c])   (weights -> Bt layout)
__global__ void transpose_cvt_kernel(const float* __restrict__ in, int rows, int cols,
                                     bf16_t* __restrict__ out) {
    int i = blockIdx.x * blockDim.x + threadIdx.x;
    if (i >= rows * cols) return;
    int r = i / cols, c = i % cols;
    out[(size_t)c * rows + r] = f2bf(in[i]);
}

__global__ void scatter_edges_kernel(const int* __restrict__ rows, const int* __restrict__ cols,
                                     const float* __restrict__ vals, float* __restrict__ A, int ne) {
    int e = blockIdx.x * blockDim.x + threadIdx.x;
    if (e < ne) atomicAdd(&A[(size_t)rows[e] * NNODE + cols[e]], vals[e]);
}

__global__ void cvt_adj_kernel(const float* __restrict__ A, bf16_t* __restrict__ Ab,
                               bf16_t* __restrict__ Atb) {
    size_t idx = (size_t)blockIdx.x * blockDim.x + threadIdx.x;
    if (idx >= (size_t)NNODE * NNODE) return;
    size_t i = idx >> 12, j = idx & (NNODE - 1);
    bf16_t v = f2bf(A[idx]);
    Ab[idx] = v;
    Atb[(j << 12) + i] = v;
}

// pred = log_softmax(concat(z_iv, z_c) @ lin_W + lin_b) over 2 classes; mi = 0.
__global__ void head_kernel(const float* __restrict__ z_iv, const float* __restrict__ z_c,
                            const float* __restrict__ linW, const float* __restrict__ linB,
                            float* __restrict__ pred, float* __restrict__ mi) {
    int row = blockIdx.x * blockDim.x + threadIdx.x;
    if (row >= NNODE) return;
    if (row == 0) mi[0] = 0.0f;
    float s0 = linB[0], s1 = linB[1];
    const float* a = z_iv + (size_t)row * HID;
    const float* b = z_c  + (size_t)row * HID;
    for (int k = 0; k < HID; ++k) {
        float x = a[k];
        s0 += x * linW[k * 2 + 0];
        s1 += x * linW[k * 2 + 1];
    }
    for (int k = 0; k < HID; ++k) {
        float x = b[k];
        s0 += x * linW[(HID + k) * 2 + 0];
        s1 += x * linW[(HID + k) * 2 + 1];
    }
    float mx  = fmaxf(s0, s1);
    float lse = mx + logf(expf(s0 - mx) + expf(s1 - mx));
    pred[row * 2 + 0] = s0 - lse;
    pred[row * 2 + 1] = s1 - lse;
}

// ---------------------------------------------------------------------------
// Orchestration
// ---------------------------------------------------------------------------
extern "C" void kernel_launch(void* const* d_in, const int* in_sizes, int n_in,
                              void* d_out, int out_size, void* d_ws, size_t ws_size,
                              hipStream_t stream) {
    const float* X        = (const float*)d_in[0];
    const int*   adj_rows = (const int*)  d_in[1];
    const int*   adj_cols = (const int*)  d_in[2];
    const float* adj_vals = (const float*)d_in[3];
    const float* W_shared = (const float*)d_in[4];
    const float* Wmat[6]  = { (const float*)d_in[5], (const float*)d_in[6], (const float*)d_in[7],
                              (const float*)d_in[8], (const float*)d_in[9], (const float*)d_in[10] };
    const float* lin_W    = (const float*)d_in[11];
    const float* lin_b    = (const float*)d_in[12];
    const int NE = in_sizes[1];

    const size_t NN  = (size_t)NNODE * NNODE;     // 16,777,216
    const size_t ZSZ = (size_t)NNODE * HID;       //  1,048,576

    // ---- workspace layout -------------------------------------------------
    uint8_t* w = (uint8_t*)d_ws;
    size_t off = 0;
    auto take = [&](size_t bytes) -> void* {
        void* p = w + off;
        off = (off + bytes + 255) & ~(size_t)255;
        return p;
    };
    float*  A_f32 = (float*)take(NN * 4);          // dead after cvt_adj; recycled below
    bf16_t* A_bf  = (bf16_t*)take(NN * 2);
    bf16_t* At_bf = (bf16_t*)take(NN * 2);
    bf16_t* X_bf  = (bf16_t*)take((size_t)NNODE * FIN * 2);
    bf16_t* WshT  = (bf16_t*)take((size_t)HID * FIN * 2);
    bf16_t* WT[6];
    for (int i = 0; i < 6; ++i) WT[i] = (bf16_t*)take((size_t)HID * HID * 2);
    bf16_t* XW_t  = (bf16_t*)take(ZSZ * 2);
    bf16_t* h_bf  = (bf16_t*)take(ZSZ * 2);
    bf16_t* hW_t  = (bf16_t*)take(ZSZ * 2);
    bf16_t* hg_t  = (bf16_t*)take(ZSZ * 2);
    bf16_t* p_t   = (bf16_t*)take(ZSZ * 2);
    bf16_t* q_t   = (bf16_t*)take(ZSZ * 2);
    bf16_t* z0_bf[3]; for (int i = 0; i < 3; ++i) z0_bf[i] = (bf16_t*)take(ZSZ * 2);
    bf16_t* z_bf[3];  for (int i = 0; i < 3; ++i) z_bf[i]  = (bf16_t*)take(ZSZ * 2);
    // recycle the dead A_f32 region: S (32MB) + graphite residual u (4MB)
    bf16_t* S = (bf16_t*)A_f32;
    float*  u = (float*)((uint8_t*)A_f32 + NN * 2);

    // ---- output layout ----------------------------------------------------
    float* out = (float*)d_out;
    float* o_z[3]   = { out,                 out + ZSZ + NN,          out + 2 * (ZSZ + NN) };
    float* o_rec[3] = { out + ZSZ,           out + 2 * ZSZ + NN,      out + 3 * ZSZ + 2 * NN };
    float* o_mi     = out + 3 * (ZSZ + NN);
    float* o_pred   = o_mi + 1;

    auto gemm = [&](const bf16_t* Aop, const bf16_t* Bt, int M, int N, int K,
                    const float* res, int act, float* oF, bf16_t* oB, bf16_t* oBT) {
        dim3 grid(N / BN, M / BM);
        wgemm_nt_bf16<<<grid, 256, 0, stream>>>(Aop, Bt, M, N, K, res, act, oF, oB, oBT);
    };

    // ---- 1) prep: casts, transposed weights, dense adjacency --------------
    cvt_bf16_kernel<<<(unsigned)((NNODE * FIN + 255) / 256), 256, 0, stream>>>(X, X_bf, (size_t)NNODE * FIN);
    transpose_cvt_kernel<<<(FIN * HID + 255) / 256, 256, 0, stream>>>(W_shared, FIN, HID, WshT);
    for (int i = 0; i < 6; ++i)
        transpose_cvt_kernel<<<(HID * HID + 255) / 256, 256, 0, stream>>>(Wmat[i], HID, HID, WT[i]);
    zero_f32_kernel<<<(unsigned)((NN + 255) / 256), 256, 0, stream>>>(A_f32, NN);
    scatter_edges_kernel<<<(NE + 255) / 256, 256, 0, stream>>>(adj_rows, adj_cols, adj_vals, A_f32, NE);
    cvt_adj_kernel<<<(unsigned)((NN + 255) / 256), 256, 0, stream>>>(A_f32, A_bf, At_bf);

    // ---- 2) h = relu(A @ (X @ Wsh)) ---------------------------------------
    gemm(X_bf, WshT, NNODE, HID, FIN,   nullptr, 0, nullptr, nullptr, XW_t);
    gemm(A_bf, XW_t, NNODE, HID, NNODE, nullptr, 1, nullptr, h_bf,   nullptr);

    // ---- 3) z0_t = relu(A @ (h @ W_t)), t in {iv, c, r} -------------------
    for (int t = 0; t < 3; ++t) {
        gemm(h_bf, WT[t], NNODE, HID, HID,   nullptr, 0, nullptr, nullptr,  hW_t);
        gemm(A_bf, hW_t,  NNODE, HID, NNODE, nullptr, 1, nullptr, z0_bf[t], nullptr);
    }

    // ---- 4) graphite per branch; S = sigmoid(z0 z0^T) is symmetric --------
    for (int t = 0; t < 3; ++t) {
        gemm(z0_bf[t], z0_bf[t], NNODE, NNODE, HID, nullptr, 2, nullptr, S, nullptr);   // S
        gemm(z0_bf[t], WT[3 + t], NNODE, HID, HID,  nullptr, 0, nullptr, nullptr, hg_t);// hg
        gemm(At_bf, hg_t, NNODE, HID, NNODE, nullptr, 0, nullptr, nullptr, p_t);        // p = A^T hg
        gemm(S,     hg_t, NNODE, HID, NNODE, nullptr, 0, nullptr, nullptr, q_t);        // q = S hg
        gemm(A_bf,  p_t,  NNODE, HID, NNODE, nullptr, 0, u, nullptr, nullptr);          // u = A p
        gemm(S,     q_t,  NNODE, HID, NNODE, u,       1, o_z[t], z_bf[t], nullptr);     // z = relu(Sq+u)
        gemm(z_bf[t], z_bf[t], NNODE, NNODE, HID, nullptr, 0, o_rec[t], nullptr, nullptr); // z z^T
    }

    // ---- 5) classifier head + mi scalar -----------------------------------
    head_kernel<<<NNODE / 256, 256, 0, stream>>>(o_z[0], o_z[1], lin_W, lin_b, o_pred, o_mi);
}
